// DiffusionMultiHeadAttention_89970974917005
// MI455X (gfx1250) — matrix operand order repaired and checked by reading, not compiled, not run on hardware
//
#include <hip/hip_runtime.h>
#include <hip/hip_bf16.h>
#include <math.h>

// ---------------------------------------------------------------------------
// DiffusionMultiHeadAttention for MI455X (gfx1250, wave32, WMMA).
// Heavy matmuls: v_wmma_f32_16x16x32_bf16 (fp32 accumulate).
// Attention K/V tiles staged via GLOBAL_LOAD_ASYNC_TO_LDS_B128 (ASYNCcnt).
// ---------------------------------------------------------------------------

#define Bc  4
#define Hc  16
#define Sc  1024
#define Dc  1024
#define DKc 64

typedef __attribute__((ext_vector_type(16))) __bf16 v16bf;
typedef __attribute__((ext_vector_type(8)))  float  v8f;

__device__ __forceinline__ v8f wmma_bf16(v16bf a, v16bf b, v8f c) {
  // (neg_a, A, neg_b, B, c_mod, C, reuse_a, reuse_b)
  return __builtin_amdgcn_wmma_f32_16x16x32_bf16(false, a, false, b, (short)0, c,
                                                 false, false);
}

__device__ __forceinline__ v8f v8f_zero() {
  v8f z;
#pragma unroll
  for (int e = 0; e < 8; ++e) z[e] = 0.0f;
  return z;
}

// A fragment: 16(M) x 32(K) bf16. Lane holds row M = lane%16.
// lanes<16 hold K 0..15, lanes>=16 hold K 16..31 -> K = 16*half + e.
__device__ __forceinline__ v16bf load_a_frag(const __bf16* rowp, int half) {
  v16bf a;
#pragma unroll
  for (int e = 0; e < 16; ++e) {
    int k = 16 * half + e;
    a[e] = rowp[k];
  }
  return a;
}

// B fragment: 32(K) x 16(N) bf16. Lane holds col N = lane%16;
// lanes<16 hold K 0..15, lanes>=16 hold K 16..31 -> K = 16*half + e.
__device__ __forceinline__ v16bf load_b_frag(const __bf16* colp, int half) {
  v16bf b;
#pragma unroll
  for (int e = 0; e < 16; ++e) b[e] = colp[16 * half + e];
  return b;
}

// Async copy of one 16B chunk global -> LDS (CDNA5, ASYNCcnt-tracked).
// ldsoff = wave-relative LDS byte address (low 32 bits of generic LDS ptr).
__device__ __forceinline__ void async_g2l_b128(unsigned ldsoff, const void* g) {
  unsigned long long ga = (unsigned long long)(uintptr_t)g;
  asm volatile("global_load_async_to_lds_b128 %0, %1, off"
               :: "v"(ldsoff), "v"(ga)
               : "memory");
}
__device__ __forceinline__ void wait_async0() {
  asm volatile("s_wait_asynccnt 0" ::: "memory");
}

// ---------------------------------------------------------------------------
// Tiled GEMM: Y = X @ W + bias.  M=4096, N=1024, K=1024.
// Block tile 128x128, K-step 32, 8 waves (wave = 32x64 output, 8 C-frags).
// AMODE: 0 = A fp32 global, 1 = A bf16 global (fp32->bf16 converted in LDS).
// OMODE: 0 = fp32 row-major (final output)
//        1 = bf16 (b,h,s,dk) layout  (Q, K)
//        2 = bf16 (b,h,dk,s) layout  (V transposed for PV WMMA)
// ---------------------------------------------------------------------------
template <int AMODE, int OMODE>
__global__ __launch_bounds__(256) void gemm_kernel(const void* __restrict__ Aptr,
                                                   const float* __restrict__ W,
                                                   const float* __restrict__ bias,
                                                   void* __restrict__ Out,
                                                   int M, int N, int K) {
  __shared__ __bf16 As[128][32];
  __shared__ __bf16 Bs[128][34];

  const int t    = threadIdx.x;
  const int lane = t & 31;
  const int w    = t >> 5;
  const int half = lane >> 4;
  const int l16  = lane & 15;
  const int n0   = blockIdx.x * 128;
  const int m0   = blockIdx.y * 128;
  const int wm   = w & 3;   // 4 M-subtiles of 32 rows
  const int wn   = w >> 2;  // 2 N-subtiles of 64 cols

  const float*  Af = (const float*)Aptr;
  const __bf16* Ab = (const __bf16*)Aptr;

  v8f acc[2][4];
#pragma unroll
  for (int i = 0; i < 2; ++i)
#pragma unroll
    for (int j = 0; j < 4; ++j) acc[i][j] = v8f_zero();

  for (int k0 = 0; k0 < K; k0 += 32) {
    // Stage A tile (128x32) into LDS as bf16.
#pragma unroll
    for (int i = 0; i < 16; ++i) {
      int e  = t + i * 256;
      int mi = e >> 5, ki = e & 31;
      size_t gi = (size_t)(m0 + mi) * K + (k0 + ki);
      float v;
      if constexpr (AMODE == 0) v = Af[gi];
      else                      v = (float)Ab[gi];
      As[mi][ki] = (__bf16)v;
    }
    // Stage B tile (32x128) into LDS transposed: Bs[n][k] so contraction pairs
    // are contiguous for B-fragment reads.
#pragma unroll
    for (int i = 0; i < 16; ++i) {
      int e  = t + i * 256;
      int kk = e >> 7, nn = e & 127;
      Bs[nn][kk] = (__bf16)W[(size_t)(k0 + kk) * N + (n0 + nn)];
    }
    __syncthreads();

    v16bf afr[2], bfr[4];
#pragma unroll
    for (int mi2 = 0; mi2 < 2; ++mi2)
      afr[mi2] = load_a_frag(&As[wm * 32 + mi2 * 16 + l16][0], half);
#pragma unroll
    for (int ni = 0; ni < 4; ++ni)
      bfr[ni] = load_b_frag(&Bs[wn * 64 + ni * 16 + l16][0], half);

#pragma unroll
    for (int mi2 = 0; mi2 < 2; ++mi2)
#pragma unroll
      for (int ni = 0; ni < 4; ++ni)
        acc[mi2][ni] = wmma_bf16(afr[mi2], bfr[ni], acc[mi2][ni]);

    __syncthreads();
  }

  // Epilogue: bias + store per C-frag layout (row = j + 8*half, col = lane%16).
#pragma unroll
  for (int mi2 = 0; mi2 < 2; ++mi2)
#pragma unroll
    for (int ni = 0; ni < 4; ++ni)
#pragma unroll
      for (int j = 0; j < 8; ++j) {
        int m = m0 + wm * 32 + mi2 * 16 + j + 8 * half;
        int n = n0 + wn * 64 + ni * 16 + l16;
        float v = acc[mi2][ni][j] + bias[n];
        if constexpr (OMODE == 0) {
          ((float*)Out)[(size_t)m * N + n] = v;
        } else if constexpr (OMODE == 1) {
          int bb = m >> 10, s = m & 1023, h = n >> 6, d = n & 63;
          ((__bf16*)Out)[(((size_t)(bb * Hc + h)) * Sc + s) * DKc + d] = (__bf16)v;
        } else {
          int bb = m >> 10, s = m & 1023, h = n >> 6, d = n & 63;
          ((__bf16*)Out)[(((size_t)(bb * Hc + h)) * DKc + d) * Sc + s] = (__bf16)v;
        }
      }
}

// ---------------------------------------------------------------------------
// Time predictor: per (b,h) compute mean/max/softmax-variance stats of
// s = clip(QK^T/sqrt(dk), +-50) with WMMA + online softmax, then 3->16->1 MLP.
// Grid: 64 blocks (one per b,h) x 256 threads (8 waves x 16 query rows).
// ---------------------------------------------------------------------------
__global__ __launch_bounds__(256) void time_pred_kernel(
    const __bf16* __restrict__ Qb, const __bf16* __restrict__ Kb,
    const float* __restrict__ Wt1, const float* __restrict__ bt1,
    const float* __restrict__ Wt2, const float* __restrict__ bt2,
    float* __restrict__ tout) {
  const int bh   = blockIdx.x;
  const int t    = threadIdx.x;
  const int lane = t & 31;
  const int w    = t >> 5;
  const int half = lane >> 4;
  const int l16  = lane & 15;

  const __bf16* Qp = Qb + (size_t)bh * Sc * DKc;
  const __bf16* Kp = Kb + (size_t)bh * Sc * DKc;

  __shared__ float part[16][3];

  float wclip = 0.0f, wrmax = 0.0f, wvar = 0.0f;

  for (int it = 0; it < 8; ++it) {
    const int q0 = (w * 8 + it) * 16;
    v16bf aq0 = load_a_frag(Qp + (size_t)(q0 + l16) * DKc, half);
    v16bf aq1 = load_a_frag(Qp + (size_t)(q0 + l16) * DKc + 32, half);

    float sumc[8], rmax[8], em[8], eZ[8], eZ2[8];
#pragma unroll
    for (int j = 0; j < 8; ++j) {
      sumc[j] = 0.0f; rmax[j] = -INFINITY; em[j] = -INFINITY;
      eZ[j] = 0.0f; eZ2[j] = 0.0f;
    }

    for (int kt = 0; kt < 64; ++kt) {
      v16bf bk0 = load_b_frag(Kp + (size_t)(kt * 16 + l16) * DKc, half);
      v16bf bk1 = load_b_frag(Kp + (size_t)(kt * 16 + l16) * DKc + 32, half);
      v8f c = v8f_zero();
      c = wmma_bf16(aq0, bk0, c);
      c = wmma_bf16(aq1, bk1, c);
#pragma unroll
      for (int j = 0; j < 8; ++j) {
        float s = c[j] * 0.125f;                 // 1/sqrt(64)
        s = fminf(fmaxf(s, -50.0f), 50.0f);
        sumc[j] += s;
        rmax[j] = fmaxf(rmax[j], s);
        float x    = 0.5f * s;                   // softmax(s/2)
        float nm   = fmaxf(em[j], x);
        float a    = __expf(em[j] - nm);
        float bfac = __expf(x - nm);
        eZ[j]  = eZ[j]  * a     + bfac;
        eZ2[j] = eZ2[j] * a * a + bfac * bfac;
        em[j]  = nm;
      }
    }

    // Reduce the 16 column-stripes of each row (butterfly within half-wave).
#pragma unroll
    for (int mask = 1; mask <= 8; mask <<= 1) {
#pragma unroll
      for (int j = 0; j < 8; ++j) {
        sumc[j] += __shfl_xor(sumc[j], mask, 32);
        rmax[j]  = fmaxf(rmax[j], __shfl_xor(rmax[j], mask, 32));
        float om  = __shfl_xor(em[j], mask, 32);
        float oZ  = __shfl_xor(eZ[j], mask, 32);
        float oZ2 = __shfl_xor(eZ2[j], mask, 32);
        float nm = fmaxf(em[j], om);
        float a  = __expf(em[j] - nm);
        float b2 = __expf(om - nm);
        eZ[j]  = eZ[j]  * a     + oZ  * b2;
        eZ2[j] = eZ2[j] * a * a + oZ2 * b2 * b2;
        em[j]  = nm;
      }
    }

    if (l16 == 0) {
#pragma unroll
      for (int j = 0; j < 8; ++j) {
        wclip += sumc[j];
        wrmax += rmax[j];
        float var = (eZ2[j] / (eZ[j] * eZ[j]) - 1.0f / (float)Sc) /
                    ((float)Sc - 1.0f);
        wvar += var;
      }
    }
  }

  if (l16 == 0) {
    part[w * 2 + half][0] = wclip;
    part[w * 2 + half][1] = wrmax;
    part[w * 2 + half][2] = wvar;
  }
  __syncthreads();

  if (t == 0) {
    float tc = 0.0f, tm = 0.0f, te = 0.0f;
    for (int i = 0; i < 16; ++i) { tc += part[i][0]; tm += part[i][1]; te += part[i][2]; }
    float mean_sim = tc / ((float)Sc * (float)Sc);
    float max_sim  = tm / (float)Sc;
    float ent      = te / (float)Sc;
    mean_sim = (mean_sim == mean_sim) ? mean_sim : 0.0f;
    max_sim  = (max_sim  == max_sim)  ? max_sim  : 0.0f;
    ent      = (ent      == ent)      ? ent      : 0.1f;
    mean_sim = fminf(fmaxf(mean_sim, -10.0f), 10.0f);
    max_sim  = fminf(fmaxf(max_sim,  -10.0f), 10.0f);
    ent      = fminf(fmaxf(ent, 0.0f), 1.0f);

    // MLP: tanh(feats @ Wt1(3x16) + bt1) @ Wt2(16x1) + bt2
    float raw = bt2[0];
    for (int i = 0; i < 16; ++i) {
      float h = bt1[i] + mean_sim * Wt1[i] + max_sim * Wt1[16 + i] + ent * Wt1[32 + i];
      raw += tanhf(h) * Wt2[i];
    }
    float tt = 0.01f + (1.0f / (1.0f + __expf(-raw))) * (2.0f - 0.01f);
    tt = fminf(fmaxf(tt, 0.01f), 2.0f);
    tt = fminf(tt, 0.85f);
    tout[bh] = tt;
  }
}

// ---------------------------------------------------------------------------
// Flash attention: out(b,s,h*dk) = softmax(QK^T/(2t)) @ V.
// Grid: (B*H)*8 blocks; block = 8 waves x 16 query rows (128-row q-block).
// K/V tiles are staged once per block via async global->LDS DMA and shared by
// all 8 waves (8x reduction of L2->WGP traffic vs per-wave fragment loads).
// ---------------------------------------------------------------------------
__global__ __launch_bounds__(256) void attn_kernel(
    const __bf16* __restrict__ Qb, const __bf16* __restrict__ Kb,
    const __bf16* __restrict__ Vt, const float* __restrict__ tvals,
    __bf16* __restrict__ aout) {
  const int blk = blockIdx.x;
  const int bh  = blk >> 3;
  const int qb  = blk & 7;
  const int bq  = bh >> 4;   // batch
  const int hh  = bh & 15;   // head

  const int t    = threadIdx.x;
  const int lane = t & 31;
  const int w    = t >> 5;
  const int half = lane >> 4;
  const int l16  = lane & 15;

  const __bf16* Qp = Qb + (size_t)bh * Sc * DKc;
  const __bf16* Kp = Kb + (size_t)bh * Sc * DKc;
  const __bf16* Vp = Vt + (size_t)bh * DKc * Sc;  // [dk][s]

  const float scale = 1.0f / (2.0f * tvals[bh]);

  __shared__ __bf16 Ks[32][72];     // 32 keys x 64 dk (+8 pad) : rows 144B
  __shared__ __bf16 Vs[64][40];     // 64 dk  x 32 keys (+8 pad): rows 80B
  __shared__ __bf16 ps[8][16][34];  // per-wave P tile (16 q x 32 k)

  // LDS byte offsets (low 32 bits of a generic LDS pointer = LDS address).
  const unsigned ksBase = (unsigned)(uintptr_t)(void*)&Ks[0][0];
  const unsigned vsBase = (unsigned)(uintptr_t)(void*)&Vs[0][0];
  // K tile: 256 threads x 16B; key = t/8, 16B-seg = t%8 (row = 128B of data).
  const int kKey = t >> 3, kSeg = t & 7;
  const unsigned ldsK = ksBase + (unsigned)(kKey * 144 + kSeg * 16);
  // V tile: row d = t/4, 16B-seg = t%4 (row = 64B of data).
  const int vRow = t >> 2, vSeg = t & 3;
  const unsigned ldsV = vsBase + (unsigned)(vRow * 80 + vSeg * 16);

  const int q0 = qb * 128 + w * 16;
  v16bf aq0 = load_a_frag(Qp + (size_t)(q0 + l16) * DKc, half);
  v16bf aq1 = load_a_frag(Qp + (size_t)(q0 + l16) * DKc + 32, half);

  v8f o[4];
#pragma unroll
  for (int f = 0; f < 4; ++f) o[f] = v8f_zero();
  float rm[8], rz[8];
#pragma unroll
  for (int j = 0; j < 8; ++j) { rm[j] = -INFINITY; rz[j] = 0.0f; }

  for (int kt = 0; kt < 32; ++kt) {
    const int k0 = kt * 32;

    // ---- stage K (32x64) and V (64x32) tiles: async global -> LDS ----
    __syncthreads();  // previous iteration's LDS reads complete
    async_g2l_b128(ldsK, (const char*)(Kp + (size_t)(k0 + kKey) * DKc) + kSeg * 16);
    async_g2l_b128(ldsV, (const char*)(Vp + (size_t)vRow * Sc + k0) + vSeg * 16);
    wait_async0();
    __syncthreads();  // tiles visible to all waves

    // ---- scores for 16q x 32k ----
    v8f sc[2];
#pragma unroll
    for (int kb2 = 0; kb2 < 2; ++kb2) {
      v16bf b0 = load_b_frag(&Ks[kb2 * 16 + l16][0], half);
      v16bf b1 = load_b_frag(&Ks[kb2 * 16 + l16][32], half);
      v8f c = v8f_zero();
      c = wmma_bf16(aq0, b0, c);
      c = wmma_bf16(aq1, b1, c);
#pragma unroll
      for (int j = 0; j < 8; ++j) c[j] *= scale;
      sc[kb2] = c;
    }

    // ---- online softmax update ----
    float corr[8];
#pragma unroll
    for (int j = 0; j < 8; ++j) {
      float tmax = fmaxf(sc[0][j], sc[1][j]);
#pragma unroll
      for (int mask = 1; mask <= 8; mask <<= 1)
        tmax = fmaxf(tmax, __shfl_xor(tmax, mask, 32));
      float nm = fmaxf(rm[j], tmax);
      corr[j] = __expf(rm[j] - nm);
      rm[j] = nm;
    }
#pragma unroll
    for (int kb2 = 0; kb2 < 2; ++kb2)
#pragma unroll
      for (int j = 0; j < 8; ++j)
        sc[kb2][j] = __expf(sc[kb2][j] - rm[j]);
#pragma unroll
    for (int j = 0; j < 8; ++j) {
      float ls = sc[0][j] + sc[1][j];
#pragma unroll
      for (int mask = 1; mask <= 8; mask <<= 1)
        ls += __shfl_xor(ls, mask, 32);
      rz[j] = rz[j] * corr[j] + ls;
    }
#pragma unroll
    for (int f = 0; f < 4; ++f)
#pragma unroll
      for (int j = 0; j < 8; ++j) o[f][j] *= corr[j];

    // ---- P tile: C-layout -> LDS -> A-layout fragment ----
#pragma unroll
    for (int kb2 = 0; kb2 < 2; ++kb2)
#pragma unroll
      for (int j = 0; j < 8; ++j)
        ps[w][j + 8 * half][kb2 * 16 + l16] = (__bf16)sc[kb2][j];
    __syncthreads();

    v16bf pa = load_a_frag(&ps[w][l16][0], half);
#pragma unroll
    for (int db = 0; db < 4; ++db) {
      v16bf bv = load_b_frag(&Vs[db * 16 + l16][0], half);
      o[db] = wmma_bf16(pa, bv, o[db]);
    }
  }

  // normalize and store to (b, s, h*dk) bf16 for the output GEMM
  float invz[8];
#pragma unroll
  for (int j = 0; j < 8; ++j) {
    float z = rz[j];
    invz[j] = (z > 0.0f) ? (1.0f / z) : 0.0f;  // nan_to_num
  }
#pragma unroll
  for (int db = 0; db < 4; ++db)
#pragma unroll
    for (int j = 0; j < 8; ++j) {
      int q = q0 + j + 8 * half;
      int d = db * 16 + l16;
      aout[((size_t)bq * Sc + q) * Dc + hh * DKc + d] = (__bf16)(o[db][j] * invz[j]);
    }
}

// ---------------------------------------------------------------------------
extern "C" void kernel_launch(void* const* d_in, const int* in_sizes, int n_in,
                              void* d_out, int out_size, void* d_ws, size_t ws_size,
                              hipStream_t stream) {
  (void)in_sizes; (void)n_in; (void)out_size; (void)ws_size;

  const float* query = (const float*)d_in[0];
  const float* key_  = (const float*)d_in[1];
  const float* value = (const float*)d_in[2];
  const float* Wq = (const float*)d_in[3];  const float* bq = (const float*)d_in[4];
  const float* Wk = (const float*)d_in[5];  const float* bk = (const float*)d_in[6];
  const float* Wv = (const float*)d_in[7];  const float* bv = (const float*)d_in[8];
  const float* Wo = (const float*)d_in[9];  const float* bo = (const float*)d_in[10];
  const float* Wt1 = (const float*)d_in[11]; const float* bt1 = (const float*)d_in[12];
  const float* Wt2 = (const float*)d_in[13]; const float* bt2 = (const float*)d_in[14];

  char* ws = (char*)d_ws;
  size_t off = 0;
  auto carve = [&](size_t bytes) {
    void* p = (void*)(ws + off);
    off = (off + bytes + 255) & ~(size_t)255;
    return p;
  };
  const size_t nElem = (size_t)Bc * Hc * Sc * DKc;     // 4M elements
  __bf16* Qb  = (__bf16*)carve(nElem * 2);             // (b,h,s,dk)
  __bf16* Kb  = (__bf16*)carve(nElem * 2);             // (b,h,s,dk)
  __bf16* Vt  = (__bf16*)carve(nElem * 2);             // (b,h,dk,s)
  __bf16* Ab  = (__bf16*)carve(nElem * 2);             // (b,s,h*dk)
  float*  tv  = (float*)carve(Bc * Hc * sizeof(float));

  const int M = Bc * Sc, N = Dc, K = Dc;
  dim3 gg(N / 128, M / 128);  // (8, 32)

  gemm_kernel<0, 1><<<gg, 256, 0, stream>>>(query, Wq, bq, (void*)Qb, M, N, K);
  gemm_kernel<0, 1><<<gg, 256, 0, stream>>>(key_,  Wk, bk, (void*)Kb, M, N, K);
  gemm_kernel<0, 2><<<gg, 256, 0, stream>>>(value, Wv, bv, (void*)Vt, M, N, K);

  time_pred_kernel<<<Bc * Hc, 256, 0, stream>>>(Qb, Kb, Wt1, bt1, Wt2, bt2, tv);

  attn_kernel<<<Bc * Hc * 8, 256, 0, stream>>>(Qb, Kb, Vt, tv, Ab);

  gemm_kernel<1, 0><<<gg, 256, 0, stream>>>(Ab, Wo, bo, d_out, M, N, K);
}
